// EGNNLayer_58875411693658
// MI455X (gfx1250) — matrix operand order repaired and checked by reading, not compiled, not run on hardware
//
#include <hip/hip_runtime.h>

typedef _Float16 v16h __attribute__((ext_vector_type(16)));
typedef _Float16 v8h  __attribute__((ext_vector_type(8)));
typedef _Float16 v4h  __attribute__((ext_vector_type(4)));
typedef float    v8f  __attribute__((ext_vector_type(8)));
typedef float    v4f  __attribute__((ext_vector_type(4)));

#define HDIM 128
#define NT   8            // 128/16 N-tiles
#define FEAT_STRIDE 264   // 256 K + pad (halves); 528B row, 16B-aligned
#define BUF_STRIDE  136   // 128 K + pad (halves); 272B row, 16B-aligned

__device__ __forceinline__ float silu_f(float x) {
    return x / (1.0f + __expf(-x));
}

// A fragment (16x32 f16, MxK) from LDS row-major [row][K].
// lanes 0-15 = rows 0-15 holding K {0..7, 16..23}; lanes 16-31 = K {8..15, 24..31}.
__device__ __forceinline__ v16h load_a_lds(const _Float16* base, int stride,
                                           int kbase, int lane) {
    int row = lane & 15;
    int kh  = (lane >> 4) & 1;
    const _Float16* p = base + row * stride + kbase + kh * 8;
    v8h lo = *(const v8h*)(p);
    v8h hi = *(const v8h*)(p + 16);
    v16h a;
#pragma unroll
    for (int q = 0; q < 8; ++q) { a[q] = lo[q]; a[8 + q] = hi[q]; }
    return a;
}

// B fragment (32x16 f16, KxN) from pre-packed weights:
// packed[(frag*32 + lane)*16 + i] == W[K = ktile*32 + (lane>>4)*8 + (i&7) + (i>>3)*16][N = jtile*16 + (lane&15)]
__device__ __forceinline__ v16h load_b_packed(const _Float16* packed, int frag, int lane) {
    return *(const v16h*)(packed + ((long)frag * 32 + lane) * 16);
}

// C fragment (16x16 f32): VGPR i, lane -> row = i + 8*(lane>>4), col = lane&15.
__device__ __forceinline__ void store_c_silu_f16(_Float16* base, int stride,
                                                 int jbase, int lane, v8f c) {
    int n  = lane & 15;
    int kh = (lane >> 4) & 1;
#pragma unroll
    for (int i = 0; i < 8; ++i) {
        base[(i + 8 * kh) * stride + jbase + n] = (_Float16)silu_f(c[i]);
    }
}

// Packed f16 atomic add (GLOBAL_ATOMIC_PK_ADD_F16, no return -> STOREcnt).
__device__ __forceinline__ void atomic_pk_add_f16(unsigned long long addr,
                                                  unsigned int data) {
    asm volatile("global_atomic_pk_add_f16 %0, %1, off"
                 :: "v"(addr), "v"(data) : "memory");
}

// ---------------- weight pre-pack / small utility kernels ----------------

__global__ void zero_f32_k(float* __restrict__ p, long n) {
    long t = (long)blockIdx.x * blockDim.x + threadIdx.x;
    if (t < n) p[t] = 0.0f;
}

__global__ void cvt_vec_k(const float* __restrict__ src, _Float16* __restrict__ dst, int n) {
    int t = blockIdx.x * blockDim.x + threadIdx.x;
    if (t < n) dst[t] = (_Float16)src[t];
}

// Pack fp32 weight [K][128] (row-major, K multiple of 32) into B-fragment layout f16.
__global__ void pack_b_k(const float* __restrict__ src, _Float16* __restrict__ dst, int K) {
    int t = blockIdx.x * blockDim.x + threadIdx.x;
    int total = K * HDIM;
    if (t >= total) return;
    int i    = t & 15;
    int lane = (t >> 4) & 31;
    int frag = t >> 9;
    int j = frag & (NT - 1);
    int k = frag >> 3;
    int srcK = k * 32 + ((lane >> 4) << 3) + (i & 7) + ((i >> 3) << 4);
    int srcN = j * 16 + (lane & 15);
    dst[t] = (_Float16)src[srcK * HDIM + srcN];
}

// ---------------- edge kernel: 32 edges / wave, single-wave workgroup ----------------

__global__ __launch_bounds__(32) void egnn_edge_k(
    const float* __restrict__ h, const float* __restrict__ pos,
    const int* __restrict__ ei,
    const _Float16* __restrict__ eW0p, const _Float16* __restrict__ eW0r,
    const float* __restrict__ eb0,
    const _Float16* __restrict__ eW1p, const float* __restrict__ eb1,
    const _Float16* __restrict__ cW0p, const float* __restrict__ cb0,
    const _Float16* __restrict__ cW1h,
    _Float16* __restrict__ m_ih, float* __restrict__ pos_acc,
    float* __restrict__ degp, int Nn, int Ee)
{
    __shared__ __align__(16) _Float16 featS[32][FEAT_STRIDE];
    __shared__ __align__(16) _Float16 b0S[32][BUF_STRIDE];
    __shared__ __align__(16) _Float16 b1S[32][BUF_STRIDE];
    __shared__ int   rowS[32], colS[32];
    __shared__ float radS[32], cdS[32][3], cwS[32];

    const int  lane = threadIdx.x & 31;
    const long e0   = (long)blockIdx.x * 32;
    if (e0 >= Ee) return;   // single-wave workgroup: uniform exit is safe

    __builtin_prefetch(eW0p, 0, 0);

    // --- per-edge metadata: one edge per lane ---
    {
        long ed = e0 + lane;
        if (ed >= Ee) ed = Ee - 1;            // clamp; side effects guarded later
        int r = ei[ed], c = ei[(long)Ee + ed];
        rowS[lane] = r;  colS[lane] = c;
        float dx = pos[3 * r + 0] - pos[3 * c + 0];
        float dy = pos[3 * r + 1] - pos[3 * c + 1];
        float dz = pos[3 * r + 2] - pos[3 * c + 2];
        cdS[lane][0] = dx; cdS[lane][1] = dy; cdS[lane][2] = dz;
        radS[lane] = dx * dx + dy * dy + dz * dz;
    }
    __syncthreads();

    // --- gather h[row] || h[col] -> LDS f16 [32][256] ---
#pragma unroll 4
    for (int e = 0; e < 32; ++e) {
        int r = rowS[e], c = colS[e];
        v4f a = ((const v4f*)(h + (long)r * HDIM))[lane];
        v4f b = ((const v4f*)(h + (long)c * HDIM))[lane];
        int kb = lane * 4;
        v4h fa, fb;
#pragma unroll
        for (int q = 0; q < 4; ++q) { fa[q] = (_Float16)a[q]; fb[q] = (_Float16)b[q]; }
        *(v4h*)&featS[e][kb]       = fa;
        *(v4h*)&featS[e][128 + kb] = fb;
    }
    __syncthreads();

    const int n  = lane & 15;
    const int kh = (lane >> 4) & 1;

    // --- GEMM1: [32x257] @ eW0 -> SiLU (radial column folded as rank-1 into acc) ---
    for (int j = 0; j < NT; ++j) {
        float bias = eb0[j * 16 + n];
        float w256 = (float)eW0r[j * 16 + n];
        v8f acc0, acc1;
#pragma unroll
        for (int i = 0; i < 8; ++i) {
            acc0[i] = bias + radS[i + 8 * kh] * w256;
            acc1[i] = bias + radS[16 + i + 8 * kh] * w256;
        }
#pragma unroll
        for (int k = 0; k < 8; ++k) {
            v16h b  = load_b_packed(eW0p, k * NT + j, lane);
            v16h a0 = load_a_lds(&featS[0][0],  FEAT_STRIDE, k * 32, lane);
            v16h a1 = load_a_lds(&featS[16][0], FEAT_STRIDE, k * 32, lane);
            acc0 = __builtin_amdgcn_wmma_f32_16x16x32_f16(false, a0, false, b,
                                                          (short)0, acc0, false, false);
            acc1 = __builtin_amdgcn_wmma_f32_16x16x32_f16(false, a1, false, b,
                                                          (short)0, acc1, false, false);
        }
        store_c_silu_f16(&b0S[0][0],  BUF_STRIDE, j * 16, lane, acc0);
        store_c_silu_f16(&b0S[16][0], BUF_STRIDE, j * 16, lane, acc1);
    }
    __syncthreads();

    // --- GEMM2: @ eW1 -> SiLU = m_ij ---
    for (int j = 0; j < NT; ++j) {
        float bias = eb1[j * 16 + n];
        v8f acc0, acc1;
#pragma unroll
        for (int i = 0; i < 8; ++i) { acc0[i] = bias; acc1[i] = bias; }
#pragma unroll
        for (int k = 0; k < 4; ++k) {
            v16h b  = load_b_packed(eW1p, k * NT + j, lane);
            v16h a0 = load_a_lds(&b0S[0][0],  BUF_STRIDE, k * 32, lane);
            v16h a1 = load_a_lds(&b0S[16][0], BUF_STRIDE, k * 32, lane);
            acc0 = __builtin_amdgcn_wmma_f32_16x16x32_f16(false, a0, false, b,
                                                          (short)0, acc0, false, false);
            acc1 = __builtin_amdgcn_wmma_f32_16x16x32_f16(false, a1, false, b,
                                                          (short)0, acc1, false, false);
        }
        store_c_silu_f16(&b1S[0][0],  BUF_STRIDE, j * 16, lane, acc0);
        store_c_silu_f16(&b1S[16][0], BUF_STRIDE, j * 16, lane, acc1);
    }
    __syncthreads();

    // --- coord MLP: silu(m_ij @ cW0 + cb0) @ cW1 -> scalar per edge ---
    float part0[8], part1[8];
#pragma unroll
    for (int i = 0; i < 8; ++i) { part0[i] = 0.0f; part1[i] = 0.0f; }
    for (int j = 0; j < NT; ++j) {
        float bias = cb0[j * 16 + n];
        v8f acc0, acc1;
#pragma unroll
        for (int i = 0; i < 8; ++i) { acc0[i] = bias; acc1[i] = bias; }
#pragma unroll
        for (int k = 0; k < 4; ++k) {
            v16h b  = load_b_packed(cW0p, k * NT + j, lane);
            v16h a0 = load_a_lds(&b1S[0][0],  BUF_STRIDE, k * 32, lane);
            v16h a1 = load_a_lds(&b1S[16][0], BUF_STRIDE, k * 32, lane);
            acc0 = __builtin_amdgcn_wmma_f32_16x16x32_f16(false, a0, false, b,
                                                          (short)0, acc0, false, false);
            acc1 = __builtin_amdgcn_wmma_f32_16x16x32_f16(false, a1, false, b,
                                                          (short)0, acc1, false, false);
        }
        float w1 = (float)cW1h[j * 16 + n];
#pragma unroll
        for (int i = 0; i < 8; ++i) {
            part0[i] += silu_f(acc0[i]) * w1;
            part1[i] += silu_f(acc1[i]) * w1;
        }
    }
#pragma unroll
    for (int m = 1; m < 16; m <<= 1) {
#pragma unroll
        for (int i = 0; i < 8; ++i) {
            part0[i] += __shfl_xor(part0[i], m, 32);
            part1[i] += __shfl_xor(part1[i], m, 32);
        }
    }
    if ((lane & 15) == 0) {
#pragma unroll
        for (int i = 0; i < 8; ++i) {
            cwS[i + 8 * kh]      = part0[i];
            cwS[16 + i + 8 * kh] = part1[i];
        }
    }
    __syncthreads();

    // --- scatter: coord update + degree (one edge per lane) ---
    if ((e0 + lane) < Ee) {
        int r = rowS[lane];
        float inv = cwS[lane] * rsqrtf(radS[lane] + 1e-8f);
        atomicAdd(&pos_acc[3 * r + 0], cdS[lane][0] * inv);
        atomicAdd(&pos_acc[3 * r + 1], cdS[lane][1] * inv);
        atomicAdd(&pos_acc[3 * r + 2], cdS[lane][2] * inv);
        atomicAdd(&degp[r], 1.0f);
    }
    // --- scatter: m_ij -> m_i via packed f16 atomics (2 per lane per edge) ---
    for (int e = 0; e < 32; ++e) {
        if ((e0 + e) >= Ee) break;
        int r = rowS[e];
        const unsigned int* src32 = (const unsigned int*)&b1S[e][0];
        unsigned int d0 = src32[lane * 2 + 0];
        unsigned int d1 = src32[lane * 2 + 1];
        unsigned long long base =
            (unsigned long long)(m_ih + (long)r * HDIM + lane * 4);
        atomic_pk_add_f16(base,     d0);
        atomic_pk_add_f16(base + 4, d1);
    }
}

// ---------------- node kernel: node MLP + residual + pos update ----------------

__global__ __launch_bounds__(64) void egnn_node_k(
    const float* __restrict__ h, const float* __restrict__ pos,
    const _Float16* __restrict__ nW0p, const float* __restrict__ nb0,
    const _Float16* __restrict__ nW1p, const float* __restrict__ nb1,
    const _Float16* __restrict__ m_ih, const float* __restrict__ pos_acc,
    const float* __restrict__ degp,
    float* __restrict__ h_out, float* __restrict__ pos_out, int Nn)
{
    __shared__ __align__(16) _Float16 featS[2][16][FEAT_STRIDE];
    __shared__ __align__(16) _Float16 b0S[2][16][BUF_STRIDE];

    const int wv   = threadIdx.x >> 5;
    const int lane = threadIdx.x & 31;
    const long n0  = ((long)blockIdx.x * 2 + wv) * 16;
    const bool active = (n0 < Nn);

    // --- gather [h || m_i] -> LDS f16 [16][256] (m_i already f16) ---
#pragma unroll 4
    for (int e = 0; e < 16; ++e) {
        long node = n0 + e;
        if (node >= Nn) node = Nn - 1;
        v4f a = ((const v4f*)(h    + node * HDIM))[lane];
        v4h b = ((const v4h*)(m_ih + node * HDIM))[lane];
        int kb = lane * 4;
        v4h fa;
#pragma unroll
        for (int q = 0; q < 4; ++q) fa[q] = (_Float16)a[q];
        *(v4h*)&featS[wv][e][kb]       = fa;
        *(v4h*)&featS[wv][e][128 + kb] = b;
    }
    __syncthreads();

    const int n  = lane & 15;
    const int kh = (lane >> 4) & 1;

    // --- GEMM1: [16x256] @ nW0 + nb0 -> SiLU ---
    for (int j = 0; j < NT; ++j) {
        float bias = nb0[j * 16 + n];
        v8f acc;
#pragma unroll
        for (int i = 0; i < 8; ++i) acc[i] = bias;
#pragma unroll
        for (int k = 0; k < 8; ++k) {
            v16h a = load_a_lds(&featS[wv][0][0], FEAT_STRIDE, k * 32, lane);
            v16h b = load_b_packed(nW0p, k * NT + j, lane);
            acc = __builtin_amdgcn_wmma_f32_16x16x32_f16(false, a, false, b,
                                                         (short)0, acc, false, false);
        }
        store_c_silu_f16(&b0S[wv][0][0], BUF_STRIDE, j * 16, lane, acc);
    }
    __syncthreads();

    // --- GEMM2: @ nW1 + nb1 + residual h -> h_out ---
    for (int j = 0; j < NT; ++j) {
        v8f acc;
#pragma unroll
        for (int i = 0; i < 8; ++i) acc[i] = nb1[j * 16 + n];
#pragma unroll
        for (int k = 0; k < 4; ++k) {
            v16h a = load_a_lds(&b0S[wv][0][0], BUF_STRIDE, k * 32, lane);
            v16h b = load_b_packed(nW1p, k * NT + j, lane);
            acc = __builtin_amdgcn_wmma_f32_16x16x32_f16(false, a, false, b,
                                                         (short)0, acc, false, false);
        }
        if (active) {
#pragma unroll
            for (int i = 0; i < 8; ++i) {
                long node = n0 + i + 8 * kh;
                if (node < Nn) {
                    long idx = node * HDIM + j * 16 + n;
                    h_out[idx] = h[idx] + acc[i];
                }
            }
        }
    }

    // --- pos update ---
    if (active && lane < 16) {
        long node = n0 + lane;
        if (node < Nn) {
            float d = degp[node] + 1e-6f;
#pragma unroll
            for (int c = 0; c < 3; ++c) {
                pos_out[node * 3 + c] = pos[node * 3 + c] + pos_acc[node * 3 + c] / d;
            }
        }
    }
}

// ---------------- launcher ----------------

extern "C" void kernel_launch(void* const* d_in, const int* in_sizes, int n_in,
                              void* d_out, int out_size, void* d_ws, size_t ws_size,
                              hipStream_t stream)
{
    const float* h   = (const float*)d_in[0];
    const float* pos = (const float*)d_in[1];
    const int*   ei  = (const int*)d_in[2];
    const float* eW0 = (const float*)d_in[3];
    const float* eb0 = (const float*)d_in[4];
    const float* eW1 = (const float*)d_in[5];
    const float* eb1 = (const float*)d_in[6];
    const float* nW0 = (const float*)d_in[7];
    const float* nb0 = (const float*)d_in[8];
    const float* nW1 = (const float*)d_in[9];
    const float* nb1 = (const float*)d_in[10];
    const float* cW0 = (const float*)d_in[11];
    const float* cb0 = (const float*)d_in[12];
    const float* cW1 = (const float*)d_in[13];

    const int Nn = in_sizes[0] / HDIM;   // 50000
    const int Ee = in_sizes[2] / 2;      // 800000

    char* ws = (char*)d_ws;
    size_t o = 0;
    _Float16* eW0p = (_Float16*)(ws + o); o += (size_t)256 * HDIM * 2;
    _Float16* eW1p = (_Float16*)(ws + o); o += (size_t)128 * HDIM * 2;
    _Float16* nW0p = (_Float16*)(ws + o); o += (size_t)256 * HDIM * 2;
    _Float16* nW1p = (_Float16*)(ws + o); o += (size_t)128 * HDIM * 2;
    _Float16* cW0p = (_Float16*)(ws + o); o += (size_t)128 * HDIM * 2;
    _Float16* eW0r = (_Float16*)(ws + o); o += (size_t)HDIM * 2;
    _Float16* cW1h = (_Float16*)(ws + o); o += (size_t)HDIM * 2;
    o = (o + 255) & ~(size_t)255;
    _Float16* m_ih   = (_Float16*)(ws + o); o += (size_t)Nn * HDIM * 2;  // f16 accumulator
    float*    pos_acc = (float*)(ws + o);   o += (size_t)Nn * 3 * 4;
    float*    degp    = (float*)(ws + o);   o += (size_t)Nn * 4;
    // m_ih, pos_acc, degp contiguous -> single zero pass (f16 zero == 0x0000)
    long zcount = (long)Nn * (HDIM / 2 + 3 + 1);

    float* h_out   = (float*)d_out;
    float* pos_out = h_out + (long)Nn * HDIM;

    zero_f32_k<<<(unsigned)((zcount + 255) / 256), 256, 0, stream>>>((float*)m_ih, zcount);

    cvt_vec_k<<<1, 128, 0, stream>>>(eW0 + (size_t)256 * HDIM, eW0r, HDIM);
    cvt_vec_k<<<1, 128, 0, stream>>>(cW1, cW1h, HDIM);

    pack_b_k<<<(256 * HDIM + 255) / 256, 256, 0, stream>>>(eW0, eW0p, 256);
    pack_b_k<<<(128 * HDIM + 255) / 256, 256, 0, stream>>>(eW1, eW1p, 128);
    pack_b_k<<<(256 * HDIM + 255) / 256, 256, 0, stream>>>(nW0, nW0p, 256);
    pack_b_k<<<(128 * HDIM + 255) / 256, 256, 0, stream>>>(nW1, nW1p, 128);
    pack_b_k<<<(128 * HDIM + 255) / 256, 256, 0, stream>>>(cW0, cW0p, 128);

    egnn_edge_k<<<(unsigned)((Ee + 31) / 32), 32, 0, stream>>>(
        h, pos, ei, eW0p, eW0r, eb0, eW1p, eb1, cW0p, cb0, cW1h,
        m_ih, pos_acc, degp, Nn, Ee);

    egnn_node_k<<<(unsigned)((Nn + 31) / 32), 64, 0, stream>>>(
        h, pos, nW0p, nb0, nW1p, nb1, m_ih, pos_acc, degp,
        h_out, pos_out, Nn);
}
